// GConv2_43748536877664
// MI455X (gfx1250) — compile-verified, matrix operand order checked by reading, not run on hardware
//
#include <hip/hip_runtime.h>
#include <hip/hip_bf16.h>

#define B_ 8
#define S_ 4096
#define D_ 768
#define K_ 64
#define TD_ 1536      // 2*D
#define DK_ 49152     // D*K

typedef __attribute__((ext_vector_type(8)))  _Float16 h8;
typedef __attribute__((ext_vector_type(16))) _Float16 v16h;
typedef __attribute__((ext_vector_type(8)))  float    v8f;

__device__ __forceinline__ float sigmoidf_(float x) {
    return 1.0f / (1.0f + __expf(-x));
}

// ---- mean pool: partial sums (deterministic two-pass, no float atomics) ----
__global__ void __launch_bounds__(256) mean_partial(const float* __restrict__ x,
                                                    float* __restrict__ part) {
    int d = blockIdx.x * 256 + threadIdx.x;     // 0..767 (grid.x = 3)
    int b = blockIdx.y;                         // 0..7
    int z = blockIdx.z;                         // 0..31 -> 128 rows each
    const float* p = x + (size_t)b * S_ * D_ + (size_t)(z * 128) * D_ + d;
    float a = 0.f;
    for (int i = 0; i < 128; ++i) a += p[(size_t)i * D_];
    part[(size_t)z * (B_ * D_) + b * D_ + d] = a;
}

__global__ void __launch_bounds__(256) mean_reduce(const float* __restrict__ part,
                                                   float* __restrict__ xg) {
    int i = blockIdx.x * 256 + threadIdx.x;     // 0..6143
    float a = 0.f;
    for (int z = 0; z < 32; ++z) a += part[(size_t)z * (B_ * D_) + i];
    xg[i] = a * (1.0f / (float)S_);
}

// ---- h = silu(xg @ W1 + b1) : [8,1536] ----
__global__ void __launch_bounds__(256) mlp1(const float* __restrict__ xg,
                                            const float* __restrict__ W1,
                                            const float* __restrict__ b1,
                                            float* __restrict__ h) {
    __shared__ float sx[D_];
    int b = blockIdx.y;
    int n = blockIdx.x * 256 + threadIdx.x;     // grid.x = 6
    for (int j = threadIdx.x; j < D_; j += 256) sx[j] = xg[b * D_ + j];
    __syncthreads();
    float a = b1[n];
    for (int k = 0; k < D_; ++k) a = fmaf(sx[k], W1[(size_t)k * TD_ + n], a);
    h[b * TD_ + n] = a * sigmoidf_(a);          // silu
}

// ---- kern = h @ W2 + b2 : [8, 49152]; streams the 302MB W2 exactly once ----
__global__ void __launch_bounds__(128) mlp2(const float* __restrict__ h,
                                            const float* __restrict__ W2,
                                            const float* __restrict__ b2,
                                            float* __restrict__ kern) {
    __shared__ float sh[B_ * TD_];              // 48 KB
    int n = blockIdx.x * 128 + threadIdx.x;     // grid.x = 384
    for (int j = threadIdx.x; j < B_ * TD_; j += 128) sh[j] = h[j];
    __syncthreads();
    float acc[B_];
#pragma unroll
    for (int b = 0; b < B_; ++b) acc[b] = 0.f;
    for (int k = 0; k < TD_; ++k) {
        float w = W2[(size_t)k * DK_ + n];      // coalesced
#pragma unroll
        for (int b = 0; b < B_; ++b) acc[b] = fmaf(sh[b * TD_ + k], w, acc[b]);
    }
    float bb = b2[n];
#pragma unroll
    for (int b = 0; b < B_; ++b) kern[(size_t)b * DK_ + n] = acc[b] + bb;
}

// ---- Wct[n][k] = (f16) Wc[k][n] : B-fragments become contiguous 16B loads ----
__global__ void __launch_bounds__(256) prep_wct(const float* __restrict__ Wc,
                                                _Float16* __restrict__ wct) {
    int i = blockIdx.x * 256 + threadIdx.x;     // grid.x = 2304 (768*768/256)
    int n = i / D_;
    int k = i - n * D_;
    wct[i] = (_Float16)Wc[(size_t)k * D_ + n];
}

// ---- fused: gate GEMM (WMMA f16->f32) + causal depthwise conv + multiply ----
// block = (b, 32-row s-tile); 256 threads = 8 waves; each wave owns 96 N-cols.
__global__ void __launch_bounds__(256) fused_gate_conv(const float* __restrict__ x,
                                                       const _Float16* __restrict__ wct,
                                                       const float* __restrict__ bc,
                                                       const float* __restrict__ kern,
                                                       float* __restrict__ out) {
    __shared__ __align__(16) _Float16 sA[32 * 776];   // 48.5 KB; reused for gate
    const int b   = blockIdx.y;
    const int s0  = blockIdx.x * 32;
    const int tid = threadIdx.x;
    const float* xb = x + (size_t)b * S_ * D_;

    // stage x tile f32 -> f16 (coalesced reads, contiguous LDS writes)
#pragma unroll 8
    for (int j = 0; j < 96; ++j) {
        int idx = tid + 256 * j;
        int r = idx / D_;
        int c = idx - r * D_;
        sA[r * 776 + c] = (_Float16)xb[(size_t)(s0 + r) * D_ + c];
    }
    __syncthreads();

    const int l    = tid & 31;
    const int wv   = tid >> 5;
    const int l15  = l & 15;
    const int kb   = (l & 16) ? 8 : 0;    // K sub-group per half-wave (A/B layout)
    const int moff = (l & 16) ? 8 : 0;    // C/D row offset per half-wave
    const int n0   = wv * 96;

    v8f acc[2][6];
#pragma unroll
    for (int mt = 0; mt < 2; ++mt)
#pragma unroll
        for (int nt = 0; nt < 6; ++nt) acc[mt][nt] = v8f{};

    for (int kt = 0; kt < 24; ++kt) {
        const int k0 = kt * 32;
        v16h af[2];
#pragma unroll
        for (int mt = 0; mt < 2; ++mt) {
            int row = mt * 16 + l15;
            h8 lo = *(const h8*)&sA[row * 776 + k0 + kb];
            h8 hi = *(const h8*)&sA[row * 776 + k0 + kb + 16];
#pragma unroll
            for (int i = 0; i < 8; ++i) { af[mt][i] = lo[i]; af[mt][i + 8] = hi[i]; }
        }
#pragma unroll
        for (int nt = 0; nt < 6; ++nt) {
            int col = n0 + nt * 16 + l15;
            const _Float16* wp = wct + (size_t)col * D_ + k0 + kb;
            h8 lo = *(const h8*)wp;
            h8 hi = *(const h8*)(wp + 16);
            v16h bf;
#pragma unroll
            for (int i = 0; i < 8; ++i) { bf[i] = lo[i]; bf[i + 8] = hi[i]; }
            acc[0][nt] = __builtin_amdgcn_wmma_f32_16x16x32_f16(
                false, af[0], false, bf, (short)0, acc[0][nt], false, false);
            acc[1][nt] = __builtin_amdgcn_wmma_f32_16x16x32_f16(
                false, af[1], false, bf, (short)0, acc[1][nt], false, false);
        }
    }
    __syncthreads();   // everyone done reading sA (A-tiles)

    // gate = sigmoid(acc + bc) -> LDS (aliases sA), f16
#pragma unroll
    for (int nt = 0; nt < 6; ++nt) {
        int col = n0 + nt * 16 + l15;
        float bcv = bc[col];
#pragma unroll
        for (int mt = 0; mt < 2; ++mt) {
#pragma unroll
            for (int r = 0; r < 8; ++r) {
                int row = mt * 16 + moff + r;
                sA[row * 776 + col] = (_Float16)sigmoidf_(acc[mt][nt][r] + bcv);
            }
        }
    }
    __syncthreads();

    // causal depthwise conv (length 64) + gate multiply; x window from L2
    for (int c = 0; c < 3; ++c) {
        const int d = tid + 256 * c;
        const float* kp = kern + (size_t)b * DK_ + (size_t)d * K_;
        float w[64];
#pragma unroll
        for (int i = 0; i < 16; ++i) {
            float4 v4 = *(const float4*)(kp + i * 4);
            w[i * 4 + 0] = v4.x; w[i * 4 + 1] = v4.y;
            w[i * 4 + 2] = v4.z; w[i * 4 + 3] = v4.w;
        }
        float xv[95];                           // rows s0-63 .. s0+31
#pragma unroll
        for (int i = 0; i < 95; ++i) {
            int row = s0 - 63 + i;
            xv[i] = (row >= 0) ? xb[(size_t)row * D_ + d] : 0.0f;
        }
#pragma unroll
        for (int s = 0; s < 32; ++s) {
            float a0 = 0.f, a1 = 0.f;
#pragma unroll
            for (int k = 0; k < 64; k += 2) {
                a0 = fmaf(w[k],     xv[s + 63 - k], a0);
                a1 = fmaf(w[k + 1], xv[s + 62 - k], a1);
            }
            float g = (float)sA[s * 776 + d];
            out[(size_t)b * S_ * D_ + (size_t)(s0 + s) * D_ + d] = (a0 + a1) * g;
        }
    }
}

extern "C" void kernel_launch(void* const* d_in, const int* in_sizes, int n_in,
                              void* d_out, int out_size, void* d_ws, size_t ws_size,
                              hipStream_t stream) {
    const float* x  = (const float*)d_in[0];
    const float* W1 = (const float*)d_in[1];
    const float* b1 = (const float*)d_in[2];
    const float* W2 = (const float*)d_in[3];
    const float* b2 = (const float*)d_in[4];
    const float* Wc = (const float*)d_in[5];
    const float* bc = (const float*)d_in[6];
    float* out = (float*)d_out;

    // workspace layout (floats), all 16B aligned
    float* ws   = (float*)d_ws;
    float* xg   = ws;                                   //   6144
    float* h    = xg   + B_ * D_;                       //  12288
    float* kern = h    + B_ * TD_;                      // 393216
    float* part = kern + (size_t)B_ * DK_;              // 196608
    _Float16* wct = (_Float16*)(part + 32 * B_ * D_);   // 589824 halves

    prep_wct    <<<dim3((D_ * D_) / 256), 256, 0, stream>>>(Wc, wct);
    mean_partial<<<dim3(3, 8, 32),        256, 0, stream>>>(x, part);
    mean_reduce <<<dim3((B_ * D_) / 256), 256, 0, stream>>>(part, xg);
    mlp1        <<<dim3(TD_ / 256, B_),   256, 0, stream>>>(xg, W1, b1, h);
    mlp2        <<<dim3(DK_ / 128),       128, 0, stream>>>(h, W2, b2, kern);
    fused_gate_conv<<<dim3(S_ / 32, B_),  256, 0, stream>>>(x, wct, bc, kern, out);
}
